// LLaDA_18519898980743
// MI455X (gfx1250) — matrix operand order repaired and checked by reading, not compiled, not run on hardware
//
#include <hip/hip_runtime.h>

#define LAYERS 12
#define DMODEL 768
#define NHEAD  12
#define HDIM   64
#define FFN    3072
#define VOCAB  50304
#define BATCH  2
#define SEQ    1024
#define TOK    (BATCH*SEQ)   // 2048 rows

typedef __attribute__((ext_vector_type(16))) __bf16 v16bf;
typedef __attribute__((ext_vector_type(8)))  __bf16 v8bf;
typedef __attribute__((ext_vector_type(8)))  float  v8f;
typedef __attribute__((ext_vector_type(4)))  unsigned int v4u;
typedef __attribute__((ext_vector_type(8)))  int v8i;
typedef __attribute__((ext_vector_type(4)))  int v4i;

union BFrag { v16bf v; v8bf h[2]; };

__device__ __forceinline__ v8f wmma_bf16(const BFrag& a, const BFrag& b, v8f c) {
  // D = A(16x32 bf16) * B(32x16 bf16) + C(16x16 f32)
  return __builtin_amdgcn_wmma_f32_16x16x32_bf16(false, a.v, false, b.v,
                                                 (short)0, c, false, false);
}

// Wave32 DPP row (16-lane) butterfly reduction: pure VALU, no LDS traffic.
#define FROR(v, N) __int_as_float(__builtin_amdgcn_mov_dpp( \
    __float_as_int(v), 0x120 + (N), 0xF, 0xF, false))
__device__ __forceinline__ float rowmax16(float v) {
  v = fmaxf(v, FROR(v, 8)); v = fmaxf(v, FROR(v, 4));
  v = fmaxf(v, FROR(v, 2)); v = fmaxf(v, FROR(v, 1));
  return v;
}
__device__ __forceinline__ float rowsum16(float v) {
  v += FROR(v, 8); v += FROR(v, 4); v += FROR(v, 2); v += FROR(v, 1);
  return v;
}

#if __has_builtin(__builtin_amdgcn_tensor_load_to_lds) && \
    __has_builtin(__builtin_amdgcn_s_wait_tensorcnt)
#define HAVE_TDM 1
#else
#define HAVE_TDM 0
#endif

// ---------------------------------------------------------------------------
// Generic bf16 WMMA GEMM: C[M,N] = A[M,K](bf16) * Bt[N,K](bf16) (+ residual)
// Block tile 128x128, BK=32, 256 threads = 8 waves (4 x 2), wave tile 32x64.
// ---------------------------------------------------------------------------
#define BM  128
#define BN  128
#define BKK 32
#define LDA 40   // 32 + 8 pad; row stride 80B keeps 16B alignment
#define LDB 40

#if HAVE_TDM
// TDM: load a (rows x 32) bf16 tile, row stride = stride_elems, into LDS with
// 16B padding after each 64B row (matches LDA/LDB = 40-element stride).
__device__ __forceinline__ void tdm_load_tile32(
    unsigned lds_off, const __bf16* g, int rows, unsigned long long stride_elems)
{
  unsigned long long ga = (unsigned long long)(size_t)(const void*)g;
  v4u g0 = { 1u,                                   // count=1, user descriptor
             lds_off,                              // LDS byte address
             (unsigned)ga,                         // global_addr[31:0]
             (unsigned)((ga >> 32) & 0x01FFFFFFull) | (2u << 30) }; // addr[56:32] | type=2
  v8i g1 = {
    (1 << 16) | (1 << 20) | (3 << 22) | (3 << 25), // data_size=2B, pad_en, every 16 DW pad 4 DW
    (32 & 0xFFFF) << 16,                           // abar=0 | tensor_dim0=32 (low16)
    ((rows & 0xFFFF) << 16),                       // tensor_dim0 hi | tensor_dim1 lo
    ((rows >> 16) & 0xFFFF) | (32 << 16),          // tensor_dim1 hi | tile_dim0=32
    (rows & 0xFFFF),                               // tile_dim1=rows, tile_dim2=0
    (int)(stride_elems & 0xFFFFFFFFull),           // tensor_dim0_stride lo32
    (int)((stride_elems >> 32) & 0xFFFFull),       // stride hi16 | dim1_stride lo16=0
    0 };
  v4i z4 = { 0, 0, 0, 0 };
  v8i z8 = { 0, 0, 0, 0, 0, 0, 0, 0 };
  __builtin_amdgcn_tensor_load_to_lds(g0, g1, z4, z4, z8, 0);
}

template<bool ADD_RES, bool OUT_F32, bool OUT_BF16>
__global__ __launch_bounds__(256) void gemm_bf16_k(
    const __bf16* __restrict__ A, const __bf16* __restrict__ Bt,
    const float* __restrict__ R, float* __restrict__ Cf, __bf16* __restrict__ Cb,
    int M, int N, int K)
{
  __shared__ __align__(16) __bf16 As[2][BM * LDA];
  __shared__ __align__(16) __bf16 Bs[2][BN * LDB];
  const int t    = threadIdx.x;
  const int lane = t & 31, wave = t >> 5;
  const int hl   = lane >> 4, l16 = lane & 15;
  const int waveM = wave >> 1, waveN = wave & 1;  // 4 x 2 wave grid
  const size_t bm = (size_t)blockIdx.y * BM;
  const size_t bn = (size_t)blockIdx.x * BN;
  const bool issuer = (wave == 0);

  const __bf16* Ab = A  + bm * (size_t)K;
  const __bf16* Bb = Bt + bn * (size_t)K;

  v8f zero = {};
  v8f acc[2][4];
  #pragma unroll
  for (int i = 0; i < 2; i++)
    #pragma unroll
    for (int j = 0; j < 4; j++) acc[i][j] = zero;

  if (issuer) {  // prologue DMA for tile 0 (per-wave op; EXEC-independent)
    tdm_load_tile32((unsigned)(size_t)(void*)&As[0][0], Ab, BM, (unsigned long long)K);
    tdm_load_tile32((unsigned)(size_t)(void*)&Bs[0][0], Bb, BN, (unsigned long long)K);
  }

  const int nIter = K / BKK;
  for (int it = 0; it < nIter; ++it) {
    const int cur = it & 1;
    __syncthreads();                       // everyone done reading buffer cur^1
    if (issuer) {
      if (it + 1 < nIter) {                // DMA next tiles, overlap with WMMAs
        tdm_load_tile32((unsigned)(size_t)(void*)&As[cur ^ 1][0],
                        Ab + (size_t)(it + 1) * BKK, BM, (unsigned long long)K);
        tdm_load_tile32((unsigned)(size_t)(void*)&Bs[cur ^ 1][0],
                        Bb + (size_t)(it + 1) * BKK, BN, (unsigned long long)K);
        __builtin_amdgcn_s_wait_tensorcnt(2);  // in-order: buffer `cur` landed
      } else {
        __builtin_amdgcn_s_wait_tensorcnt(0);
      }
    }
    __syncthreads();                       // publish buffer cur

    BFrag afr[2];
    #pragma unroll
    for (int mt = 0; mt < 2; mt++) {       // A layout: K = hl*8+e / 16+hl*8+e-8
      const __bf16* p = &As[cur][(waveM * 32 + mt * 16 + l16) * LDA];
      afr[mt].h[0] = *(const v8bf*)(p + hl * 8);
      afr[mt].h[1] = *(const v8bf*)(p + 16 + hl * 8);
    }
    #pragma unroll
    for (int nt = 0; nt < 4; nt++) {       // B layout: K = hl*8+e / 16+hl*8+e-8, N = l16
      const __bf16* p = &Bs[cur][(waveN * 64 + nt * 16 + l16) * LDB];
      BFrag bfr;
      bfr.h[0] = *(const v8bf*)(p + hl * 8);
      bfr.h[1] = *(const v8bf*)(p + 16 + hl * 8);
      #pragma unroll
      for (int mt = 0; mt < 2; mt++)
        acc[mt][nt] = wmma_bf16(afr[mt], bfr, acc[mt][nt]);
    }
  }

  #pragma unroll
  for (int mt = 0; mt < 2; mt++)
    #pragma unroll
    for (int nt = 0; nt < 4; nt++) {
      size_t col = bn + waveN * 64 + nt * 16 + l16;
      #pragma unroll
      for (int r = 0; r < 8; r++) {        // C layout: row = hl*8 + r
        size_t row = bm + waveM * 32 + mt * 16 + hl * 8 + r;
        float v = acc[mt][nt][r];
        if (ADD_RES)  v += R[row * (size_t)N + col];
        if (OUT_F32)  Cf[row * (size_t)N + col] = v;
        if (OUT_BF16) Cb[row * (size_t)N + col] = (__bf16)v;
      }
    }
}

#else  // !HAVE_TDM: vector-load fallback

template<bool ADD_RES, bool OUT_F32, bool OUT_BF16>
__global__ __launch_bounds__(256) void gemm_bf16_k(
    const __bf16* __restrict__ A, const __bf16* __restrict__ Bt,
    const float* __restrict__ R, float* __restrict__ Cf, __bf16* __restrict__ Cb,
    int M, int N, int K)
{
  __shared__ __align__(16) __bf16 As[BM * LDA];
  __shared__ __align__(16) __bf16 Bs[BN * LDB];
  const int t    = threadIdx.x;
  const int lane = t & 31, wave = t >> 5;
  const int hl   = lane >> 4, l16 = lane & 15;
  const int waveM = wave >> 1, waveN = wave & 1;
  const size_t bm = (size_t)blockIdx.y * BM;
  const size_t bn = (size_t)blockIdx.x * BN;
  const int cprow = t >> 1, cpcol = (t & 1) * 16;

  const __bf16* Ag = A  + (bm + cprow) * (size_t)K + cpcol;
  const __bf16* Bg = Bt + (bn + cprow) * (size_t)K + cpcol;
  __bf16* Asw = &As[cprow * LDA + cpcol];
  __bf16* Bsw = &Bs[cprow * LDB + cpcol];

  v8f zero = {};
  v8f acc[2][4];
  #pragma unroll
  for (int i = 0; i < 2; i++)
    #pragma unroll
    for (int j = 0; j < 4; j++) acc[i][j] = zero;

  for (int k0 = 0; k0 < K; k0 += BKK) {
    uint4 a0 = *(const uint4*)(Ag + k0);
    uint4 a1 = *(const uint4*)(Ag + k0 + 8);
    uint4 b0 = *(const uint4*)(Bg + k0);
    uint4 b1 = *(const uint4*)(Bg + k0 + 8);
    __syncthreads();
    *(uint4*)(Asw) = a0; *(uint4*)(Asw + 8) = a1;
    *(uint4*)(Bsw) = b0; *(uint4*)(Bsw + 8) = b1;
    __syncthreads();
    if (k0 + BKK < K) {
      __builtin_prefetch(Ag + k0 + BKK, 0, 0);
      __builtin_prefetch(Bg + k0 + BKK, 0, 0);
    }
    BFrag afr[2];
    #pragma unroll
    for (int mt = 0; mt < 2; mt++) {
      const __bf16* p = &As[(waveM * 32 + mt * 16 + l16) * LDA];
      afr[mt].h[0] = *(const v8bf*)(p + hl * 8);
      afr[mt].h[1] = *(const v8bf*)(p + 16 + hl * 8);
    }
    #pragma unroll
    for (int nt = 0; nt < 4; nt++) {
      const __bf16* p = &Bs[(waveN * 64 + nt * 16 + l16) * LDB];
      BFrag bfr;
      bfr.h[0] = *(const v8bf*)(p + hl * 8);
      bfr.h[1] = *(const v8bf*)(p + 16 + hl * 8);
      #pragma unroll
      for (int mt = 0; mt < 2; mt++)
        acc[mt][nt] = wmma_bf16(afr[mt], bfr, acc[mt][nt]);
    }
  }

  #pragma unroll
  for (int mt = 0; mt < 2; mt++)
    #pragma unroll
    for (int nt = 0; nt < 4; nt++) {
      size_t col = bn + waveN * 64 + nt * 16 + l16;
      #pragma unroll
      for (int r = 0; r < 8; r++) {
        size_t row = bm + waveM * 32 + mt * 16 + hl * 8 + r;
        float v = acc[mt][nt][r];
        if (ADD_RES)  v += R[row * (size_t)N + col];
        if (OUT_F32)  Cf[row * (size_t)N + col] = v;
        if (OUT_BF16) Cb[row * (size_t)N + col] = (__bf16)v;
      }
    }
}
#endif  // HAVE_TDM

// ---------------------------------------------------------------------------
// Flash attention (non-causal). Q pre-scaled by 1/sqrt(HD). 256 threads,
// 8 waves x 16 query rows = 128 q rows per block; key blocks of 64.
// ---------------------------------------------------------------------------
#define KB  64
#define LDK 72
#define LDV 72
#define LDP 72

__global__ __launch_bounds__(256) void attn_k(
    const __bf16* __restrict__ Q,  // [BH, S, HD] (scaled)
    const __bf16* __restrict__ Km, // [BH, S, HD]
    const __bf16* __restrict__ Vm, // [BH, S, HD]
    __bf16* __restrict__ O)        // [B, S, D]
{
  __shared__ __align__(16) __bf16 Kt[KB   * LDK];
  __shared__ __align__(16) __bf16 Vt[HDIM * LDV];     // transposed: [hd][key]
  __shared__ __align__(16) __bf16 Pl[8 * 16 * LDP];   // per-wave P staging

  const int t    = threadIdx.x;
  const int lane = t & 31, wave = t >> 5;
  const int hl   = lane >> 4, l16 = lane & 15;
  const int bh = blockIdx.y;
  const int b  = bh / NHEAD, h = bh % NHEAD;
  const int q0 = blockIdx.x * 128;
  const int qrow = q0 + wave * 16 + l16;

  BFrag qf[2];                       // Q rows as two A fragments (hd 0..31 / 32..63)
  const __bf16* qp = Q + ((size_t)bh * SEQ + qrow) * HDIM;
  #pragma unroll
  for (int s = 0; s < 2; s++) {
    qf[s].h[0] = *(const v8bf*)(qp + s * 32 + hl * 8);
    qf[s].h[1] = *(const v8bf*)(qp + s * 32 + 16 + hl * 8);
  }

  float m_run[8], l_run[8];
  #pragma unroll
  for (int r = 0; r < 8; r++) { m_run[r] = -1e30f; l_run[r] = 0.f; }
  v8f zero = {};
  v8f oacc[4];
  #pragma unroll
  for (int i = 0; i < 4; i++) oacc[i] = zero;

  const int cr = t >> 2, cc = (t & 3) * 16;  // 64 rows x 4 segs of 16
  __bf16* Pw = &Pl[wave * 16 * LDP];

  for (int k0 = 0; k0 < SEQ; k0 += KB) {
    __syncthreads();
    const __bf16* kg = Km + ((size_t)bh * SEQ + k0 + cr) * HDIM + cc;
    *(uint4*)(&Kt[cr * LDK + cc])     = *(const uint4*)(kg);
    *(uint4*)(&Kt[cr * LDK + cc + 8]) = *(const uint4*)(kg + 8);
    const __bf16* vg = Vm + ((size_t)bh * SEQ + k0 + cr) * HDIM + cc;
    union { uint4 u[2]; __bf16 e[16]; } vv;
    vv.u[0] = *(const uint4*)(vg);
    vv.u[1] = *(const uint4*)(vg + 8);
    #pragma unroll
    for (int i = 0; i < 16; i++) Vt[(cc + i) * LDV + cr] = vv.e[i]; // transpose
    __syncthreads();

    // scores: 4 key sub-tiles x (2 WMMAs over HD)
    v8f sc[4];
    #pragma unroll
    for (int kt = 0; kt < 4; kt++) {
      v8f c = zero;
      #pragma unroll
      for (int s = 0; s < 2; s++) {
        const __bf16* p = &Kt[(kt * 16 + l16) * LDK + s * 32];
        BFrag bf;
        bf.h[0] = *(const v8bf*)(p + hl * 8);
        bf.h[1] = *(const v8bf*)(p + 16 + hl * 8);
        c = wmma_bf16(qf[s], bf, c);
      }
      sc[kt] = c;
    }

    // online softmax: rows live in VGPR index (hl*8+r), 16 keys across lanes.
    // Cross-lane reductions via DPP row_ror butterflies (VALU only).
    #pragma unroll
    for (int r = 0; r < 8; r++) {
      float mx = fmaxf(fmaxf(sc[0][r], sc[1][r]), fmaxf(sc[2][r], sc[3][r]));
      mx = rowmax16(mx);
      float mn = fmaxf(m_run[r], mx);
      float alpha = __expf(m_run[r] - mn);
      m_run[r] = mn;
      float rs = 0.f;
      #pragma unroll
      for (int kt = 0; kt < 4; kt++) {
        float p = __expf(sc[kt][r] - mn);
        sc[kt][r] = p;
        rs += p;
      }
      rs = rowsum16(rs);
      l_run[r] = l_run[r] * alpha + rs;
      #pragma unroll
      for (int nt = 0; nt < 4; nt++) oacc[nt][r] *= alpha;
    }

    // stage P (16x64 bf16) in wave-private LDS for A-fragment reload
    #pragma unroll
    for (int kt = 0; kt < 4; kt++)
      #pragma unroll
      for (int r = 0; r < 8; r++)
        Pw[(hl * 8 + r) * LDP + kt * 16 + l16] = (__bf16)sc[kt][r];
    // cross-lane store->load within the wave: drain DS before reading back
    asm volatile("s_wait_dscnt 0" ::: "memory");

    // o += P(16x64) * V(64x64): 2 k-steps x 4 hd tiles
    #pragma unroll
    for (int ks = 0; ks < 2; ks++) {
      BFrag af;
      const __bf16* pa = &Pw[l16 * LDP + ks * 32];
      af.h[0] = *(const v8bf*)(pa + hl * 8);
      af.h[1] = *(const v8bf*)(pa + 16 + hl * 8);
      #pragma unroll
      for (int nt = 0; nt < 4; nt++) {
        const __bf16* pv = &Vt[(nt * 16 + l16) * LDV + ks * 32];
        BFrag bf;
        bf.h[0] = *(const v8bf*)(pv + hl * 8);
        bf.h[1] = *(const v8bf*)(pv + 16 + hl * 8);
        oacc[nt] = wmma_bf16(af, bf, oacc[nt]);
      }
    }
  }

  #pragma unroll
  for (int nt = 0; nt < 4; nt++)
    #pragma unroll
    for (int r = 0; r < 8; r++) {
      size_t row = q0 + wave * 16 + hl * 8 + r;
      float v = oacc[nt][r] / l_run[r];
      O[((size_t)b * SEQ + row) * DMODEL + h * HDIM + nt * 16 + l16] = (__bf16)v;
    }
}

// ---------------------------------------------------------------------------
// Elementwise / prep kernels
// ---------------------------------------------------------------------------
__global__ __launch_bounds__(256) void convT_k(  // Wt[n*K+k] = (bf16)W[k*N+n]
    const float* __restrict__ W, __bf16* __restrict__ Wt, int K, int N)
{
  __shared__ float tile[32][33];
  const int t = threadIdx.x, tx = t & 31, ty = t >> 5;
  const int n0 = blockIdx.x * 32, k0 = blockIdx.y * 32;
  #pragma unroll
  for (int i = 0; i < 4; i++)
    tile[ty + 8 * i][tx] = W[(size_t)(k0 + ty + 8 * i) * N + n0 + tx];
  __syncthreads();
  #pragma unroll
  for (int i = 0; i < 4; i++)
    Wt[(size_t)(n0 + ty + 8 * i) * K + k0 + tx] = (__bf16)tile[tx][ty + 8 * i];
}

__global__ __launch_bounds__(256) void embed_k(
    const int* __restrict__ ids, const float* __restrict__ emb, float* __restrict__ x)
{
  const int row = blockIdx.x, t = threadIdx.x;
  const float* e = emb + (size_t)ids[row] * DMODEL;
  float* xr = x + (size_t)row * DMODEL;
  xr[t] = e[t]; xr[t + 256] = e[t + 256]; xr[t + 512] = e[t + 512];
}

__global__ __launch_bounds__(256) void rmsnorm_k(
    const float* __restrict__ x, const float* __restrict__ g, __bf16* __restrict__ out)
{
  __shared__ float red[256];
  const int row = blockIdx.x, t = threadIdx.x;
  const float* xr = x + (size_t)row * DMODEL;
  float v0 = xr[t], v1 = xr[t + 256], v2 = xr[t + 512];
  red[t] = v0 * v0 + v1 * v1 + v2 * v2;
  __syncthreads();
  for (int off = 128; off > 0; off >>= 1) {
    if (t < off) red[t] += red[t + off];
    __syncthreads();
  }
  float rs = rsqrtf(red[0] * (1.f / DMODEL) + 1e-6f);
  __bf16* orow = out + (size_t)row * DMODEL;
  orow[t]       = (__bf16)(g[t] * v0 * rs);
  orow[t + 256] = (__bf16)(g[t + 256] * v1 * rs);
  orow[t + 512] = (__bf16)(g[t + 512] * v2 * rs);
}

// RoPE (full precision) + reshape [B,S,H,HD]->[B,H,S,HD] + bf16; q scaled 1/8
__global__ __launch_bounds__(256) void rope_qkv_k(
    const float* __restrict__ q, const float* __restrict__ k, const float* __restrict__ v,
    __bf16* __restrict__ qo, __bf16* __restrict__ ko, __bf16* __restrict__ vo)
{
  const int g  = blockIdx.x * 256 + threadIdx.x;
  const int d  = g % DMODEL;
  const int bs = g / DMODEL;
  const int s  = bs % SEQ;
  const int b  = bs / SEQ;
  const int h  = d >> 6;
  const int hd = d & 63;
  const int i  = hd & 31;
  float freq = __expf(-0.28782313662425572f * (float)i); // 10000^(-2i/64)
  float sn, cs;
  __sincosf((float)s * freq, &sn, &cs);
  const int pair = g + ((hd < 32) ? 32 : -32);
  const size_t o = (((size_t)b * NHEAD + h) * SEQ + s) * HDIM + hd;
  float qv = q[g], qpv = q[pair];
  float rq = (hd < 32) ? -qpv : qpv;
  qo[o] = (__bf16)((qv * cs + rq * sn) * 0.125f);
  float kv = k[g], kpv = k[pair];
  float rk = (hd < 32) ? -kpv : kpv;
  ko[o] = (__bf16)(kv * cs + rk * sn);
  vo[o] = (__bf16)v[g];
}

__global__ __launch_bounds__(256) void silu_k(
    const float* __restrict__ u, const float* __restrict__ w, __bf16* __restrict__ out)
{
  const int i = blockIdx.x * 256 + threadIdx.x;
  float x = u[i];
  out[i] = (__bf16)((x / (1.f + __expf(-x))) * w[i]);
}

// ---------------------------------------------------------------------------
// Orchestration
// ---------------------------------------------------------------------------
extern "C" void kernel_launch(void* const* d_in, const int* in_sizes, int n_in,
                              void* d_out, int out_size, void* d_ws, size_t ws_size,
                              hipStream_t stream)
{
  (void)in_sizes; (void)n_in; (void)out_size; (void)ws_size;
  const int*   ids  = (const int*)d_in[0];
  const float* emb  = (const float*)d_in[1];
  const float* Wq   = (const float*)d_in[2];
  const float* Wk   = (const float*)d_in[3];
  const float* Wv   = (const float*)d_in[4];
  const float* Wo   = (const float*)d_in[5];
  const float* g1   = (const float*)d_in[6];
  const float* W1   = (const float*)d_in[7];
  const float* W3   = (const float*)d_in[8];
  const float* W2   = (const float*)d_in[9];
  const float* g2   = (const float*)d_in[10];
  const float* gf   = (const float*)d_in[11];
  const float* Wout = (const float*)d_in[12];
  float* logits = (float*)d_out;

  char* wsb = (char*)d_ws;
  size_t off = 0;
  auto alloc = [&](size_t bytes) -> void* {
    void* p = wsb + off;
    off = (off + bytes + 255) & ~(size_t)255;
    return p;
  };

  const size_t DD = (size_t)DMODEL * DMODEL;
  const size_t DF = (size_t)DMODEL * FFN;

  __bf16* WqT  = (__bf16*)alloc(2 * LAYERS * DD);
  __bf16* WkT  = (__bf16*)alloc(2 * LAYERS * DD);
  __bf16* WvT  = (__bf16*)alloc(2 * LAYERS * DD);
  __bf16* WoT  = (__bf16*)alloc(2 * LAYERS * DD);
  __bf16* W1T  = (__bf16*)alloc(2 * LAYERS * DF);
  __bf16* W3T  = (__bf16*)alloc(2 * LAYERS * DF);
  __bf16* W2T  = (__bf16*)alloc(2 * LAYERS * DF);
  __bf16* WouT = (__bf16*)alloc(2 * (size_t)VOCAB * DMODEL);
  float*  x    = (float*) alloc(4 * (size_t)TOK * DMODEL);
  __bf16* hbuf = (__bf16*)alloc(2 * (size_t)TOK * DMODEL);
  float*  qb   = (float*) alloc(4 * (size_t)TOK * DMODEL);
  float*  kb   = (float*) alloc(4 * (size_t)TOK * DMODEL);
  float*  vb   = (float*) alloc(4 * (size_t)TOK * DMODEL);
  __bf16* qr   = (__bf16*)alloc(2 * (size_t)TOK * DMODEL);
  __bf16* kr   = (__bf16*)alloc(2 * (size_t)TOK * DMODEL);
  __bf16* vr   = (__bf16*)alloc(2 * (size_t)TOK * DMODEL);
  __bf16* ao   = (__bf16*)alloc(2 * (size_t)TOK * DMODEL);
  float*  ff1  = (float*) alloc(4 * (size_t)TOK * FFN);
  float*  ff3  = (float*) alloc(4 * (size_t)TOK * FFN);
  __bf16* ffs  = (__bf16*)alloc(2 * (size_t)TOK * FFN);

  dim3 blk(256);

  // weight convert + transpose (every call; deterministic, no caching)
  for (int l = 0; l < LAYERS; l++) {
    dim3 gDD(DMODEL / 32, DMODEL / 32);
    convT_k<<<gDD, blk, 0, stream>>>(Wq + l * DD, WqT + l * DD, DMODEL, DMODEL);
    convT_k<<<gDD, blk, 0, stream>>>(Wk + l * DD, WkT + l * DD, DMODEL, DMODEL);
    convT_k<<<gDD, blk, 0, stream>>>(Wv + l * DD, WvT + l * DD, DMODEL, DMODEL);
    convT_k<<<gDD, blk, 0, stream>>>(Wo + l * DD, WoT + l * DD, DMODEL, DMODEL);
    dim3 gDF(FFN / 32, DMODEL / 32);
    convT_k<<<gDF, blk, 0, stream>>>(W1 + l * DF, W1T + l * DF, DMODEL, FFN);
    convT_k<<<gDF, blk, 0, stream>>>(W3 + l * DF, W3T + l * DF, DMODEL, FFN);
    dim3 gFD(DMODEL / 32, FFN / 32);
    convT_k<<<gFD, blk, 0, stream>>>(W2 + l * DF, W2T + l * DF, FFN, DMODEL);
  }
  convT_k<<<dim3(VOCAB / 32, DMODEL / 32), blk, 0, stream>>>(Wout, WouT, DMODEL, VOCAB);

  embed_k<<<TOK, blk, 0, stream>>>(ids, emb, x);

  dim3 gD(DMODEL / BN, TOK / BM);
  dim3 gF(FFN / BN, TOK / BM);
  for (int l = 0; l < LAYERS; l++) {
    rmsnorm_k<<<TOK, blk, 0, stream>>>(x, g1 + (size_t)l * DMODEL, hbuf);
    gemm_bf16_k<false, true, false><<<gD, blk, 0, stream>>>(
        hbuf, WqT + l * DD, nullptr, qb, nullptr, TOK, DMODEL, DMODEL);
    gemm_bf16_k<false, true, false><<<gD, blk, 0, stream>>>(
        hbuf, WkT + l * DD, nullptr, kb, nullptr, TOK, DMODEL, DMODEL);
    gemm_bf16_k<false, true, false><<<gD, blk, 0, stream>>>(
        hbuf, WvT + l * DD, nullptr, vb, nullptr, TOK, DMODEL, DMODEL);
    rope_qkv_k<<<(TOK * DMODEL) / 256, blk, 0, stream>>>(qb, kb, vb, qr, kr, vr);
    attn_k<<<dim3(SEQ / 128, BATCH * NHEAD), blk, 0, stream>>>(qr, kr, vr, ao);
    gemm_bf16_k<true, true, false><<<gD, blk, 0, stream>>>(
        ao, WoT + l * DD, x, x, nullptr, TOK, DMODEL, DMODEL);   // x += attn@Wo
    rmsnorm_k<<<TOK, blk, 0, stream>>>(x, g2 + (size_t)l * DMODEL, hbuf);
    gemm_bf16_k<false, true, false><<<gF, blk, 0, stream>>>(
        hbuf, W1T + l * DF, nullptr, ff1, nullptr, TOK, FFN, DMODEL);
    gemm_bf16_k<false, true, false><<<gF, blk, 0, stream>>>(
        hbuf, W3T + l * DF, nullptr, ff3, nullptr, TOK, FFN, DMODEL);
    silu_k<<<(TOK * FFN) / 256, blk, 0, stream>>>(ff1, ff3, ffs);
    gemm_bf16_k<true, true, false><<<gD, blk, 0, stream>>>(
        ffs, W2T + l * DF, x, x, nullptr, TOK, DMODEL, FFN);     // x += swiglu@W2
  }

  rmsnorm_k<<<TOK, blk, 0, stream>>>(x, gf, hbuf);
  gemm_bf16_k<false, true, false><<<dim3(VOCAB / BN, TOK / BM), blk, 0, stream>>>(
      hbuf, WouT, nullptr, logits, nullptr, TOK, VOCAB, DMODEL);
}
